// NTXentLoss_37039797960862
// MI455X (gfx1250) — matrix operand order repaired and checked by reading, not compiled, not run on hardware
//
#include <hip/hip_runtime.h>
#include <hip/hip_bf16.h>
#include <math.h>

typedef __attribute__((ext_vector_type(16))) _Float16 v16h;
typedef __attribute__((ext_vector_type(8)))  _Float16 v8h;
typedef __attribute__((ext_vector_type(4)))  _Float16 v4h;
typedef __attribute__((ext_vector_type(8)))  float    v8f;

#define LOG2E    1.4426950408889634f
#define LN2      0.6931471805599453f
// rn is pre-scaled by sqrt((1/T) * log2(e)) = sqrt(14.42695...) so that the
// WMMA output is directly the exp2 argument: (s*a)·(s*b) = sim/T * log2e.
#define SCALE_H  3.7982825f

#define CH    64            // B rows staged per chunk
#define LROW  136           // padded LDS row: 68-dword stride -> conflict-free
#define NSPLIT 8            // N-sweep split across blockIdx.y (512 blocks total)

// ---------------------------------------------------------------------------
// Kernel 1: rn[row] = SCALE_H * concat(z_i,z_j)[row] / max(||row||,1e-8) (f16)
// ---------------------------------------------------------------------------
__global__ __launch_bounds__(256)
void ntx_normalize(const float* __restrict__ zi, const float* __restrict__ zj,
                   _Float16* __restrict__ rn, int N) {
  const int lane = threadIdx.x & 31;
  const int row  = blockIdx.x * 8 + (threadIdx.x >> 5);
  const float* src = (row < N) ? (zi + (size_t)row * 128)
                               : (zj + (size_t)(row - N) * 128);
  float4 v = ((const float4*)src)[lane];
  float ss = v.x * v.x + v.y * v.y + v.z * v.z + v.w * v.w;
#pragma unroll
  for (int m = 16; m >= 1; m >>= 1) ss += __shfl_xor(ss, m, 32);
  const float inv = SCALE_H / fmaxf(sqrtf(ss), 1e-8f);
  v4h h;
  h[0] = (_Float16)(v.x * inv); h[1] = (_Float16)(v.y * inv);
  h[2] = (_Float16)(v.z * inv); h[3] = (_Float16)(v.w * inv);
  *(v4h*)(rn + (size_t)row * 128 + lane * 4) = h;
}

// ---------------------------------------------------------------------------
// Kernel 2: pos[i] = dot(rn[i], rn[(i+N) mod 2N])  (carries the 14.43x scale)
// ---------------------------------------------------------------------------
__global__ __launch_bounds__(256)
void ntx_positives(const _Float16* __restrict__ rn, float* __restrict__ pos, int N) {
  const int lane = threadIdx.x & 31;
  const int row  = blockIdx.x * 8 + (threadIdx.x >> 5);
  const int prt  = (row < N) ? row + N : row - N;
  v4h a = *(const v4h*)(rn + (size_t)row * 128 + lane * 4);
  v4h b = *(const v4h*)(rn + (size_t)prt * 128 + lane * 4);
  float d = 0.f;
#pragma unroll
  for (int e = 0; e < 4; ++e) d += (float)a[e] * (float)b[e];
#pragma unroll
  for (int m = 16; m >= 1; m >>= 1) d += __shfl_xor(d, m, 32);
  if (lane == 0) pos[row] = d;
}

// ---------------------------------------------------------------------------
// Kernel 3: fused sim = rn@rn^T with rowsum_s[i] = sum_{j in split s} exp2(c).
// Per block: 8 waves x one 16-row M tile; B chunks (64 rows) staged to LDS
// (double-buffered, padded rows) so the block reads each B row from L2 once.
// Per 16x16 tile: preload all 4 B fragments (8x ds_load_b128, one dscnt wait),
// 4x back-to-back v_wmma_f32_16x16x32_f16, then 8 independent v_exp_f32
// followed by 8 independent adds (scheduler can hide TRANS latency).
// ---------------------------------------------------------------------------
__global__ __launch_bounds__(256)
void ntx_simexp(const _Float16* __restrict__ rn, float* __restrict__ rowsumP,
                int twoN) {
  __shared__ _Float16 lb[2][CH * LROW];   // 2 x 17408 B

  const int tid   = threadIdx.x;
  const int lane  = tid & 31;
  const int wave  = tid >> 5;
  const int mtile = blockIdx.x * 8 + wave;
  const int half  = lane >> 4;          // 0: lanes 0-15, 1: lanes 16-31
  const int l16   = lane & 15;

  const int colsPer = twoN / NSPLIT;    // 1024
  const int col0    = blockIdx.y * colsPer;
  const int nch     = colsPer / CH;     // 16

  // A fragments (16-bit 16x32 ISA layout), resident in VGPRs for the sweep.
  const _Float16* arow = rn + (size_t)(mtile * 16 + l16) * 128;
  v16h a[4];
#pragma unroll
  for (int kc = 0; kc < 4; ++kc) {
    const _Float16* p = arow + kc * 32 + half * 8;
    v8h lo = *(const v8h*)(p);
    v8h hi = *(const v8h*)(p + 16);
    v16h t;
#pragma unroll
    for (int e = 0; e < 8; ++e) { t[e] = lo[e]; t[e + 8] = hi[e]; }
    a[kc] = t;
  }

  float rs[8];
#pragma unroll
  for (int r = 0; r < 8; ++r) rs[r] = 0.f;

  // Staging assignment: 4 threads per B row, 64B (32 halves) each.
  const int srow  = tid >> 2;           // 0..63
  const int spart = (tid & 3) * 32;     // half offset within row

  { // stage chunk 0
    const int4* s = (const int4*)(rn + (size_t)(col0 + srow) * 128 + spart);
    int4* d = (int4*)(&lb[0][srow * LROW + spart]);
    d[0] = s[0]; d[1] = s[1]; d[2] = s[2]; d[3] = s[3];
  }
  __syncthreads();

  for (int ch = 0; ch < nch; ++ch) {
    const int buf = ch & 1;
    if (ch + 1 < nch) {   // prefetch next chunk into the other buffer
      const int4* s = (const int4*)(rn + (size_t)(col0 + (ch + 1) * CH + srow) * 128 + spart);
      int4* d = (int4*)(&lb[buf ^ 1][srow * LROW + spart]);
      d[0] = s[0]; d[1] = s[1]; d[2] = s[2]; d[3] = s[3];
    }

#pragma unroll
    for (int t = 0; t < CH / 16; ++t) {
      const _Float16* brow = &lb[buf][(t * 16 + l16) * LROW];

      // Preload all 4 K-chunk B fragments, then run the WMMA chain dry.
      v16h b[4];
#pragma unroll
      for (int kc = 0; kc < 4; ++kc) {
        // 16-bit 32x16 B layout: lanes 0-15 K=0..7,16..23; lanes 16-31 K=8..15,24..31.
        const _Float16* bp = brow + kc * 32 + half * 8;
        v8h b0 = *(const v8h*)(bp);
        v8h b1 = *(const v8h*)(bp + 16);
        v16h tb;
#pragma unroll
        for (int e = 0; e < 8; ++e) { tb[e] = b0[e]; tb[e + 8] = b1[e]; }
        b[kc] = tb;
      }

      v8f c = {0.f, 0.f, 0.f, 0.f, 0.f, 0.f, 0.f, 0.f};
#pragma unroll
      for (int kc = 0; kc < 4; ++kc)
        c = __builtin_amdgcn_wmma_f32_16x16x32_f16(
                false, a[kc], false, b[kc], (short)0, c, false, false);

      // c[r] is already sim/T * log2e -> raw v_exp_f32 (|arg| <= 14.43,
      // far from the f32 exp2 denorm cliff). Independent exps first so the
      // scheduler can overlap TRANS latency, then the adds.
      float e8[8];
#pragma unroll
      for (int r = 0; r < 8; ++r) e8[r] = __builtin_amdgcn_exp2f(c[r]);
#pragma unroll
      for (int r = 0; r < 8; ++r) rs[r] += e8[r];
    }
    __syncthreads();   // reads of buf done; writes of buf^1 done
  }

  // Reduce row partials across the 16 lanes sharing each row; deterministic
  // per-split output slot (no fp atomics -> bitwise-reproducible).
#pragma unroll
  for (int r = 0; r < 8; ++r) {
#pragma unroll
    for (int m = 8; m >= 1; m >>= 1) rs[r] += __shfl_xor(rs[r], m, 16);
  }
  if (l16 == 0) {
    float* dst = rowsumP + (size_t)blockIdx.y * twoN + mtile * 16 + half * 8;
#pragma unroll
    for (int r = 0; r < 8; ++r) dst[r] = rs[r];
  }
}

// ---------------------------------------------------------------------------
// Kernel 4: loss = mean_i [ ln(exp2(ps_i) + sum_s rowsumP[s][i]) - ps_i*ln2 ]
// ---------------------------------------------------------------------------
__global__ __launch_bounds__(256)
void ntx_loss(const float* __restrict__ pos, const float* __restrict__ rowsumP,
              float* __restrict__ out, int twoN) {
  __shared__ float sm[256];
  float acc = 0.f;
  for (int i = threadIdx.x; i < twoN; i += 256) {
    const float ps = pos[i];
    float tot = __builtin_amdgcn_exp2f(ps);
#pragma unroll
    for (int s = 0; s < NSPLIT; ++s) tot += rowsumP[(size_t)s * twoN + i];
    acc += (__builtin_amdgcn_logf(tot) - ps) * LN2;
  }
  sm[threadIdx.x] = acc;
  __syncthreads();
  for (int s = 128; s > 0; s >>= 1) {
    if ((int)threadIdx.x < s) sm[threadIdx.x] += sm[threadIdx.x + s];
    __syncthreads();
  }
  if (threadIdx.x == 0) out[0] = sm[0] / (float)twoN;
}

// ---------------------------------------------------------------------------
extern "C" void kernel_launch(void* const* d_in, const int* in_sizes, int n_in,
                              void* d_out, int out_size, void* d_ws, size_t ws_size,
                              hipStream_t stream) {
  (void)n_in; (void)out_size; (void)ws_size;
  const float* zi = (const float*)d_in[0];
  const float* zj = (const float*)d_in[1];
  const int N    = in_sizes[0] / 128;   // 4096
  const int twoN = 2 * N;               // 8192

  // Workspace: rn (2MB f16) | pos (32KB) | rowsumP (NSPLIT * 32KB)
  _Float16* rn      = (_Float16*)d_ws;
  float*    pos     = (float*)((char*)d_ws + (size_t)twoN * 128 * sizeof(_Float16));
  float*    rowsumP = pos + twoN;

  ntx_normalize<<<twoN / 8, 256, 0, stream>>>(zi, zj, rn, N);
  ntx_positives<<<twoN / 8, 256, 0, stream>>>(rn, pos, N);
  dim3 grid(twoN / 16 / 8, NSPLIT);     // 64 x 8 blocks, 8 waves each
  ntx_simexp<<<grid, 256, 0, stream>>>(rn, rowsumP, twoN);
  ntx_loss<<<1, 256, 0, stream>>>(pos, rowsumP, (float*)d_out, twoN);
}
